// GlobalFractalRouter_87368224735637
// MI455X (gfx1250) — matrix operand order repaired and checked by reading, not compiled, not run on hardware
//
#include <hip/hip_runtime.h>

#define S_LEN 2048
#define D_DIM 512
#define NB 4
#define NH 8

typedef __attribute__((ext_vector_type(16))) __bf16 v16bf;
typedef __attribute__((ext_vector_type(8))) float v8f;
typedef __attribute__((ext_vector_type(4))) unsigned int v4u;

union FRAG {
  v16bf bf;
  v4u u4[2];
};

__device__ __forceinline__ unsigned short f2bf(float f) {
  unsigned int u = __builtin_bit_cast(unsigned int, f);
  u += 0x7FFFu + ((u >> 16) & 1u);  // round-to-nearest-even
  return (unsigned short)(u >> 16);
}

__device__ __forceinline__ v8f wmma_bf16(v16bf a, v16bf b, v8f c) {
  return __builtin_amdgcn_wmma_f32_16x16x32_bf16(false, a, false, b, (short)0, c,
                                                 false, false);
}

// A-fragment slice: {p[0..15]} (caller adds +16 for lanes>=16).
__device__ __forceinline__ v16bf frag_a16(const unsigned short* p) {
  FRAG f;
  f.u4[0] = *(const v4u*)p;
  f.u4[1] = *(const v4u*)(p + 8);
  return f.bf;
}

// B-fragment from K-contiguous storage (caller adds +16 for lanes>=16).
__device__ __forceinline__ v16bf frag_b16(const unsigned short* p) {
  FRAG f;
  f.u4[0] = *(const v4u*)p;
  f.u4[1] = *(const v4u*)(p + 8);
  return f.bf;
}

// DPP16 row-rotate (within 16-lane row) — pure-VALU cross-lane reduction step.
template <int N>
__device__ __forceinline__ float row_ror(float x) {
  int i = __builtin_bit_cast(int, x);
  i = __builtin_amdgcn_update_dpp(i, i, 0x120 + N, 0xF, 0xF, false);
  return __builtin_bit_cast(float, i);
}
__device__ __forceinline__ float row_max16(float x) {
  x = fmaxf(x, row_ror<1>(x));
  x = fmaxf(x, row_ror<2>(x));
  x = fmaxf(x, row_ror<4>(x));
  x = fmaxf(x, row_ror<8>(x));
  return x;
}
__device__ __forceinline__ float row_sum16(float x) {
  x += row_ror<1>(x);
  x += row_ror<2>(x);
  x += row_ror<4>(x);
  x += row_ror<8>(x);
  return x;
}

#define ASYNC_B128(dst, off, base)                                      \
  asm volatile("global_load_async_to_lds_b128 %0, %1, %2" ::"v"(dst),   \
               "v"(off), "s"(base)                                      \
               : "memory")
#define WAIT_ASYNC() asm volatile("s_wait_asynccnt 0x0" ::: "memory")

// ---------------------------------------------------------------------------
// Fingerprint-derived small vectors: 0.1*anchor_out, score bias (+fp bias), gate
// ---------------------------------------------------------------------------
__global__ __launch_bounds__(256) void prep_kernel(
    const float* __restrict__ fp, const float* __restrict__ anchors,
    const float* __restrict__ w1, const float* __restrict__ b1,
    const float* __restrict__ w2, const float* __restrict__ b2,
    const float* __restrict__ aow, const float* __restrict__ aob,
    const float* __restrict__ score_b, const float* __restrict__ fpbw,
    const float* __restrict__ fpbb, const float* __restrict__ gw,
    const float* __restrict__ gb, float* __restrict__ anchor_comb,
    float* __restrict__ sbias, float* __restrict__ gate) {
  __shared__ float h1[32];
  __shared__ float aff[16];
  __shared__ float wv[512];
  const int t = threadIdx.x;
  if (t < 32) {
    float s = b1[t];
    for (int f = 0; f < 64; ++f) s += fp[f] * w1[f * 32 + t];
    float x3 = s * s * s;
    h1[t] = 0.5f * s * (1.0f + tanhf(0.7978845608f * (s + 0.044715f * x3)));
  }
  __syncthreads();
  if (t < 16) {
    float s = b2[t];
    for (int j = 0; j < 32; ++j) s += h1[j] * w2[j * 16 + t];
    aff[t] = 1.0f / (1.0f + __expf(-s));
  }
  __syncthreads();
  for (int d = t; d < 512; d += 256) {
    float s = 0.0f;
    for (int a = 0; a < 16; ++a) s += anchors[a * 512 + d] * aff[a];
    wv[d] = s;
  }
  __syncthreads();
  for (int d = t; d < 512; d += 256) {
    float s = aob[d];
    for (int kk = 0; kk < 512; ++kk) s += wv[kk] * aow[kk * 512 + d];
    anchor_comb[d] = 0.1f * s;
    float s2 = score_b[d] + fpbb[d];
    float s3 = gb[d];
    for (int f = 0; f < 64; ++f) {
      s2 += fp[f] * fpbw[f * 512 + d];
      s3 += fp[f] * gw[f * 512 + d];
    }
    sbias[d] = s2;
    gate[d] = 1.0f / (1.0f + __expf(-s3));
  }
}

__global__ __launch_bounds__(256) void cvt_bf16_kernel(const float* __restrict__ x,
                                                       unsigned short* __restrict__ xb,
                                                       int n) {
  int i = blockIdx.x * 256 + threadIdx.x;
  if (i < n) xb[i] = f2bf(x[i]);
}

// ---------------------------------------------------------------------------
// [8192,512]x[512,512] GEMM, bf16 pre-converted. Block tile 128x128, 8 waves
// each 32x64 (8 WMMA / 12 ds_load_b128 per k-step). Double-buffered LDS:
// async-DMA A tile and transposed W tile for step i+1 staged during step i.
// MODE 0: bf16 out head-split; MODE 1: bf16 out row-major; MODE 2: f32 combine
// ---------------------------------------------------------------------------
template <int MODE>
__global__ __launch_bounds__(256) void gemm512_kernel(
    const unsigned short* __restrict__ A, const unsigned short* __restrict__ W,
    const float* __restrict__ bias, unsigned short* __restrict__ outb,
    float* __restrict__ outf, const float* __restrict__ x,
    const float* __restrict__ anchor_comb, const float* __restrict__ gate,
    const float* __restrict__ routed) {
  __shared__ unsigned short At[2][128 * 32];  // [m][k]
  __shared__ unsigned short Wt[2][128 * 32];  // [n][k] (transposed stage)
  const int tid = threadIdx.x, lane = tid & 31, w = tid >> 5;
  const int mw = (w & 3) * 32, nw = (w >> 2) * 64;
  const int M0 = blockIdx.x * 128, N0 = blockIdx.y * 128;
  const int lrow = lane & 15;
  const int ah = (lane >= 16) ? 16 : 0;
  const int kh = (lane >= 16) ? 16 : 0;
  const unsigned ldsA = (unsigned)(size_t)(void*)At;
  const int ar = tid >> 1, ac = (tid & 1) * 16;        // A stage mapping
  const int rp = (tid & 15) * 2, cg = (tid >> 4) * 8;  // W stage mapping
  v8f acc[2][4] = {};

  auto stage_a = [&](int k0, int buf) {  // async DMA straight into LDS
    const unsigned d0 = ldsA + (unsigned)buf * 8192u + (unsigned)tid * 32u;
    const unsigned o0 = (unsigned)((M0 + ar) * 1024 + k0 * 2 + ac * 2);
    ASYNC_B128(d0, o0, A);
    ASYNC_B128(d0 + 16u, o0 + 16u, A);
  };
  auto stage_w = [&](int k0, int buf) {  // 2 x b128 loads, 8 packed b32 stores
    const v4u t0 = *(const v4u*)(W + (size_t)(k0 + rp) * 512 + N0 + cg);
    const v4u t1 = *(const v4u*)(W + (size_t)(k0 + rp + 1) * 512 + N0 + cg);
    unsigned short b0[8], b1[8];
    *(v4u*)b0 = t0;
    *(v4u*)b1 = t1;
    unsigned* wt32 = (unsigned*)Wt[buf];
#pragma unroll
    for (int j = 0; j < 8; ++j)
      wt32[((cg + j) * 32 + rp) >> 1] = (unsigned)b0[j] | ((unsigned)b1[j] << 16);
  };

  stage_a(0, 0);
  stage_w(0, 0);
  WAIT_ASYNC();
  __syncthreads();
  for (int ks = 0; ks < 16; ++ks) {
    const int cur = ks & 1, nxt = cur ^ 1;
    if (ks + 1 < 16) {  // stage next tile into the other buffer
      stage_a((ks + 1) * 32, nxt);
      stage_w((ks + 1) * 32, nxt);
    }
    const unsigned short* Ac = At[cur];
    const unsigned short* Wc = Wt[cur];
    v16bf a0 = frag_a16(Ac + (size_t)(mw + lrow) * 32 + ah);
    v16bf a1 = frag_a16(Ac + (size_t)(mw + 16 + lrow) * 32 + ah);
    v16bf bfr[4];
#pragma unroll
    for (int ni = 0; ni < 4; ++ni)
      bfr[ni] = frag_b16(Wc + (size_t)(nw + ni * 16 + lrow) * 32 + kh);
#pragma unroll
    for (int ni = 0; ni < 4; ++ni) {
      acc[0][ni] = wmma_bf16(a0, bfr[ni], acc[0][ni]);
      acc[1][ni] = wmma_bf16(a1, bfr[ni], acc[1][ni]);
    }
    WAIT_ASYNC();
    __syncthreads();
  }
  const int hi = (lane >= 16) ? 8 : 0;
#pragma unroll
  for (int mi = 0; mi < 2; ++mi)
#pragma unroll
    for (int ni = 0; ni < 4; ++ni)
#pragma unroll
      for (int r = 0; r < 8; ++r) {
        const int gm = M0 + mw + mi * 16 + r + hi;
        const int gn = N0 + nw + ni * 16 + lrow;
        const float val = acc[mi][ni][r] + bias[gn];
        if (MODE == 0) {
          const int b = gm >> 11, s = gm & 2047;
          const int h = gn >> 6, hd = gn & 63;
          outb[(((size_t)(b * NH + h)) * S_LEN + s) * 64 + hd] = f2bf(val);
        } else if (MODE == 1) {
          outb[(size_t)gm * 512 + gn] = f2bf(val);
        } else {
          const size_t o = (size_t)gm * 512 + gn;
          outf[o] = x[o] + val + anchor_comb[gn] + gate[gn] * routed[o];
        }
      }
}

// ---------------------------------------------------------------------------
// Flash attention with analytic cantor bias. 4 waves x 16 queries per block.
// Double-buffered: K tile async-DMA'd, V tile transposed (packed b32 stores)
// for chunk i+1 while chunk i computes. Row stats via DPP16 rotates.
// ---------------------------------------------------------------------------
__global__ __launch_bounds__(128) void attn_kernel(
    const unsigned short* __restrict__ q, const unsigned short* __restrict__ k,
    const unsigned short* __restrict__ v, const float* __restrict__ cscale,
    unsigned short* __restrict__ attnb) {
  __shared__ unsigned short kT[2][32 * 64];   // [key][hd]
  __shared__ unsigned short vTt[2][64 * 32];  // [hd][key]
  __shared__ unsigned short pT[4][16 * 32];
  const int tid = threadIdx.x, lane = tid & 31, w = tid >> 5;
  const int bh = blockIdx.y, h = bh & 7, b = bh >> 3;
  const int q0 = blockIdx.x * 64 + w * 16;
  const float cs = cscale[h];
  const float INV_CM = 1.0f / 2096129.0f;
  const unsigned short* qb = q + (size_t)bh * S_LEN * 64;
  const unsigned short* kb = k + (size_t)bh * S_LEN * 64;
  const unsigned short* vb = v + (size_t)bh * S_LEN * 64;
  const int lrow = lane & 15;
  const int hi = (lane >= 16) ? 8 : 0;
  const int ah = (lane >= 16) ? 16 : 0;
  const int kh = (lane >= 16) ? 16 : 0;
  const unsigned ldsK = (unsigned)(size_t)(void*)kT;
  const int vkey = (tid & 15) * 2, vhd0 = (tid >> 4) * 8;  // V stage mapping
  const v16bf qf0 = frag_a16(qb + (size_t)(q0 + lrow) * 64 + 0 + ah);
  const v16bf qf1 = frag_a16(qb + (size_t)(q0 + lrow) * 64 + 32 + ah);
  float cq[8], m[8], l[8];
  v8f acc[4] = {};
#pragma unroll
  for (int r = 0; r < 8; ++r) {
    const int qi = q0 + r + hi;
    cq[r] = (float)qi * (float)(qi + 1) * 0.5f;
    m[r] = -1e30f;
    l[r] = 0.0f;
  }

  auto stage_k = [&](int k0, int buf) {  // async DMA: 4KB K tile
    const unsigned d0 = ldsK + (unsigned)buf * 4096u + (unsigned)tid * 16u;
    const unsigned o0 = (unsigned)(k0 * 128 + tid * 16);
    ASYNC_B128(d0, o0, kb);
    ASYNC_B128(d0 + 2048u, o0 + 2048u, kb);
  };
  auto stage_v = [&](int k0, int buf) {  // key-pair x 8hd: packed b32 stores
    const v4u t0 = *(const v4u*)(vb + (size_t)(k0 + vkey) * 64 + vhd0);
    const v4u t1 = *(const v4u*)(vb + (size_t)(k0 + vkey + 1) * 64 + vhd0);
    unsigned short b0[8], b1[8];
    *(v4u*)b0 = t0;
    *(v4u*)b1 = t1;
    unsigned* vt32 = (unsigned*)vTt[buf];
#pragma unroll
    for (int j = 0; j < 8; ++j)
      vt32[((vhd0 + j) * 32 + vkey) >> 1] = (unsigned)b0[j] | ((unsigned)b1[j] << 16);
  };

  stage_k(0, 0);
  stage_v(0, 0);
  WAIT_ASYNC();
  __syncthreads();
  for (int kc = 0; kc < 64; ++kc) {
    const int cur = kc & 1, nxt = cur ^ 1;
    const int k0 = kc * 32;
    if (kc + 1 < 64) {  // stage next chunk into the other buffer
      stage_k(k0 + 32, nxt);
      stage_v(k0 + 32, nxt);
    }
    const unsigned short* kTc = kT[cur];
    const unsigned short* vTc = vTt[cur];
    v8f S0 = {}, S1 = {};
#pragma unroll
    for (int hh = 0; hh < 2; ++hh) {
      v16bf a = hh ? qf1 : qf0;
      v16bf bk0 = frag_b16(kTc + (size_t)lrow * 64 + hh * 32 + kh);
      v16bf bk1 = frag_b16(kTc + (size_t)(16 + lrow) * 64 + hh * 32 + kh);
      S0 = wmma_bf16(a, bk0, S0);
      S1 = wmma_bf16(a, bk1, S1);
    }
    const float ck0 = (float)(k0 + lrow) * (float)(k0 + lrow + 1) * 0.5f;
    const float ck1 = (float)(k0 + 16 + lrow) * (float)(k0 + 17 + lrow) * 0.5f;
#pragma unroll
    for (int r = 0; r < 8; ++r) {
      const float s0 = S0[r] * 0.125f + (1.0f - fabsf(cq[r] - ck0) * INV_CM) * cs;
      const float s1 = S1[r] * 0.125f + (1.0f - fabsf(cq[r] - ck1) * INV_CM) * cs;
      const float mx = row_max16(fmaxf(s0, s1));
      const float nm = fmaxf(m[r], mx);
      const float al = __expf(m[r] - nm);
      const float p0 = __expf(s0 - nm);
      const float p1 = __expf(s1 - nm);
      l[r] = l[r] * al + row_sum16(p0 + p1);
      m[r] = nm;
#pragma unroll
      for (int n = 0; n < 4; ++n) acc[n][r] *= al;
      pT[w][(r + hi) * 32 + lrow] = f2bf(p0);
      pT[w][(r + hi) * 32 + 16 + lrow] = f2bf(p1);
    }
    const v16bf pf = frag_a16(pT[w] + (size_t)lrow * 32 + ah);
#pragma unroll
    for (int n = 0; n < 4; ++n) {
      v16bf bv = frag_b16(vTc + (size_t)(n * 16 + lrow) * 32 + kh);
      acc[n] = wmma_bf16(pf, bv, acc[n]);
    }
    WAIT_ASYNC();
    __syncthreads();
  }
#pragma unroll
  for (int n = 0; n < 4; ++n)
#pragma unroll
    for (int r = 0; r < 8; ++r) {
      const int s = q0 + r + hi;
      const int col = h * 64 + n * 16 + lrow;
      attnb[((size_t)b * S_LEN + s) * 512 + col] = f2bf(acc[n][r] / l[r]);
    }
}

// ---------------------------------------------------------------------------
// Top-4 router: one wave per 16 queries. WMMA score tiles -> LDS -> per-lane
// running top-4 -> softmax -> gather of f32 x rows. qs staged via async DMA.
// ---------------------------------------------------------------------------
__global__ __launch_bounds__(32) void route_kernel(const unsigned short* __restrict__ qs,
                                                   const unsigned short* __restrict__ xb,
                                                   const float* __restrict__ x,
                                                   float* __restrict__ routed) {
  __shared__ unsigned short qsT[16 * 512];
  __shared__ float scoreT[16 * 16];
  __shared__ int tIdx[16 * 4];
  __shared__ float tW[16 * 4];
  const int lane = threadIdx.x;
  const int lrow = lane & 15;
  const int hi = (lane >= 16) ? 8 : 0;
  const int ah = (lane >= 16) ? 16 : 0;
  const int kh = (lane >= 16) ? 16 : 0;
  const int b = blockIdx.y;
  const int q0 = blockIdx.x * 16;
  {  // async DMA: 16KB qs tile
    const unsigned ldsQ = (unsigned)(size_t)(void*)qsT;
    const unsigned short* qbase = qs + ((size_t)b * S_LEN + q0) * 512;
    for (int i = lane; i < 1024; i += 32) {
      ASYNC_B128(ldsQ + (unsigned)i * 16u, (unsigned)i * 16u, qbase);
    }
    WAIT_ASYNC();
  }
  float tv[4] = {-1e30f, -1e30f, -1e30f, -1e30f};
  int ti[4] = {0, 0, 0, 0};
  const float scl = 0.04419417382415922f;  // 512^-0.5
  for (int kt = 0; kt < 128; ++kt) {
    v8f acc = {};
    const unsigned short* xrow = xb + ((size_t)b * S_LEN + kt * 16 + lrow) * 512;
#pragma unroll
    for (int kc = 0; kc < 16; ++kc) {
      v16bf a = frag_a16(qsT + (size_t)lrow * 512 + kc * 32 + ah);
      v16bf bb = frag_b16(xrow + kc * 32 + kh);
      acc = wmma_bf16(a, bb, acc);
    }
#pragma unroll
    for (int r = 0; r < 8; ++r) scoreT[(r + hi) * 16 + lrow] = acc[r] * scl;
    if (lane < 16) {
#pragma unroll
      for (int j = 0; j < 16; ++j) {
        const float vv = scoreT[lane * 16 + j];
        const int id = kt * 16 + j;
        if (vv > tv[0]) {
          tv[3] = tv[2]; ti[3] = ti[2]; tv[2] = tv[1]; ti[2] = ti[1];
          tv[1] = tv[0]; ti[1] = ti[0]; tv[0] = vv; ti[0] = id;
        } else if (vv > tv[1]) {
          tv[3] = tv[2]; ti[3] = ti[2]; tv[2] = tv[1]; ti[2] = ti[1];
          tv[1] = vv; ti[1] = id;
        } else if (vv > tv[2]) {
          tv[3] = tv[2]; ti[3] = ti[2]; tv[2] = vv; ti[2] = id;
        } else if (vv > tv[3]) {
          tv[3] = vv; ti[3] = id;
        }
      }
    }
  }
  if (lane < 16) {
    const float mx = tv[0];
    float e[4], sum = 0.0f;
#pragma unroll
    for (int j = 0; j < 4; ++j) { e[j] = __expf(tv[j] - mx); sum += e[j]; }
    const float inv = 1.0f / sum;
#pragma unroll
    for (int j = 0; j < 4; ++j) {
      tW[lane * 4 + j] = e[j] * inv;
      tIdx[lane * 4 + j] = ti[j];
    }
  }
  for (int qq = 0; qq < 16; ++qq) {
    const int i0 = tIdx[qq * 4 + 0], i1 = tIdx[qq * 4 + 1];
    const int i2 = tIdx[qq * 4 + 2], i3 = tIdx[qq * 4 + 3];
    const float w0 = tW[qq * 4 + 0], w1 = tW[qq * 4 + 1];
    const float w2 = tW[qq * 4 + 2], w3 = tW[qq * 4 + 3];
    const float* r0 = x + ((size_t)b * S_LEN + i0) * 512;
    const float* r1 = x + ((size_t)b * S_LEN + i1) * 512;
    const float* r2 = x + ((size_t)b * S_LEN + i2) * 512;
    const float* r3 = x + ((size_t)b * S_LEN + i3) * 512;
    float* od = routed + ((size_t)b * S_LEN + q0 + qq) * 512;
    for (int d = lane; d < 512; d += 32)
      od[d] = w0 * r0[d] + w1 * r1[d] + w2 * r2[d] + w3 * r3[d];
  }
}

extern "C" void kernel_launch(void* const* d_in, const int* in_sizes, int n_in,
                              void* d_out, int out_size, void* d_ws, size_t ws_size,
                              hipStream_t stream) {
  (void)in_sizes; (void)n_in; (void)out_size; (void)ws_size;
  const float* x            = (const float*)d_in[0];
  const float* fp           = (const float*)d_in[1];
  const float* anchors      = (const float*)d_in[2];
  const float* fp2a_w1      = (const float*)d_in[3];
  const float* fp2a_b1      = (const float*)d_in[4];
  const float* fp2a_w2      = (const float*)d_in[5];
  const float* fp2a_b2      = (const float*)d_in[6];
  const float* anchor_out_w = (const float*)d_in[7];
  const float* anchor_out_b = (const float*)d_in[8];
  const float* q_w          = (const float*)d_in[9];
  const float* q_b          = (const float*)d_in[10];
  const float* k_w          = (const float*)d_in[11];
  const float* k_b          = (const float*)d_in[12];
  const float* v_w          = (const float*)d_in[13];
  const float* v_b          = (const float*)d_in[14];
  const float* o_w          = (const float*)d_in[15];
  const float* o_b          = (const float*)d_in[16];
  const float* cantor_scale = (const float*)d_in[17];
  const float* score_w      = (const float*)d_in[18];
  const float* score_b      = (const float*)d_in[19];
  const float* fpb_w        = (const float*)d_in[20];
  const float* fpb_b        = (const float*)d_in[21];
  const float* gate_w       = (const float*)d_in[22];
  const float* gate_b       = (const float*)d_in[23];

  char* ws = (char*)d_ws;
  const size_t MB = 1024 * 1024;
  unsigned short* qB    = (unsigned short*)(ws + 0 * MB);
  unsigned short* kB    = (unsigned short*)(ws + 8 * MB);
  unsigned short* vB    = (unsigned short*)(ws + 16 * MB);
  unsigned short* qsB   = (unsigned short*)(ws + 24 * MB);
  unsigned short* xbB   = (unsigned short*)(ws + 32 * MB);
  unsigned short* attnB = (unsigned short*)(ws + 40 * MB);
  float* routed         = (float*)(ws + 48 * MB);
  float* anchor_comb    = (float*)(ws + 64 * MB);
  float* sbias          = anchor_comb + 512;
  float* gate           = anchor_comb + 1024;
  unsigned short* wqB   = (unsigned short*)(ws + 65 * MB);
  unsigned short* wkB   = wqB + 262144;
  unsigned short* wvB   = wkB + 262144;
  unsigned short* wsB   = wvB + 262144;
  unsigned short* woB   = wsB + 262144;

  prep_kernel<<<1, 256, 0, stream>>>(fp, anchors, fp2a_w1, fp2a_b1, fp2a_w2, fp2a_b2,
                                     anchor_out_w, anchor_out_b, score_b, fpb_w, fpb_b,
                                     gate_w, gate_b, anchor_comb, sbias, gate);
  const int NTOT = NB * S_LEN * D_DIM;
  cvt_bf16_kernel<<<(NTOT + 255) / 256, 256, 0, stream>>>(x, xbB, NTOT);
  const int NW = 512 * 512;
  cvt_bf16_kernel<<<(NW + 255) / 256, 256, 0, stream>>>(q_w, wqB, NW);
  cvt_bf16_kernel<<<(NW + 255) / 256, 256, 0, stream>>>(k_w, wkB, NW);
  cvt_bf16_kernel<<<(NW + 255) / 256, 256, 0, stream>>>(v_w, wvB, NW);
  cvt_bf16_kernel<<<(NW + 255) / 256, 256, 0, stream>>>(score_w, wsB, NW);
  cvt_bf16_kernel<<<(NW + 255) / 256, 256, 0, stream>>>(o_w, woB, NW);
  dim3 gg(64, 4);
  gemm512_kernel<0><<<gg, 256, 0, stream>>>(xbB, wqB, q_b, qB, nullptr, nullptr,
                                            nullptr, nullptr, nullptr);
  gemm512_kernel<0><<<gg, 256, 0, stream>>>(xbB, wkB, k_b, kB, nullptr, nullptr,
                                            nullptr, nullptr, nullptr);
  gemm512_kernel<0><<<gg, 256, 0, stream>>>(xbB, wvB, v_b, vB, nullptr, nullptr,
                                            nullptr, nullptr, nullptr);
  gemm512_kernel<1><<<gg, 256, 0, stream>>>(xbB, wsB, sbias, qsB, nullptr, nullptr,
                                            nullptr, nullptr, nullptr);
  attn_kernel<<<dim3(32, 32), 128, 0, stream>>>(qB, kB, vB, cantor_scale, attnB);
  route_kernel<<<dim3(128, 4), 32, 0, stream>>>(qsB, xbB, x, routed);
  gemm512_kernel<2><<<gg, 256, 0, stream>>>(attnB, woB, o_b, nullptr, (float*)d_out, x,
                                            anchor_comb, gate, routed);
}
